// GatedMultiHeadAttention_8194797601266
// MI455X (gfx1250) — compile-verified, matrix operand order checked by reading, not run on hardware
//
#include <hip/hip_runtime.h>

// ---------------------------------------------------------------------------
// Gated multi-head attention for MI455X (gfx1250, wave32, WMMA f16 16x16x32)
//   B=2, L=2048, D_IN=D_OUT=1024, H=16, HD=64
// Pipeline: f32->f16 converts -> fused QGKV GEMM (TDM-staged LDS tiles) ->
//           V^T build -> causal flash attention (+sigmoid gate) ->
//           output GEMM (TDM-staged, +bias, f32)
// Workspace layout (halves):
//   xh[4096*1024] | Wh[4096*1024] | Woh[1024*1024] | qgkv[4096*4096]
//   | vt[2048*2048] | ctx[4096*1024]   (total ~66 MB)
// ---------------------------------------------------------------------------

typedef _Float16 h4  __attribute__((ext_vector_type(4)));
typedef _Float16 h8  __attribute__((ext_vector_type(8)));
typedef _Float16 h16 __attribute__((ext_vector_type(16)));
typedef float    f8  __attribute__((ext_vector_type(8)));
typedef unsigned int u32x4 __attribute__((ext_vector_type(4)));
typedef int      i32x4 __attribute__((ext_vector_type(4)));
typedef int      i32x8 __attribute__((ext_vector_type(8)));

static constexpr int BB   = 2;
static constexpr int LL   = 2048;
static constexpr int DIN  = 1024;
static constexpr int DOUT = 1024;
static constexpr int HH   = 16;
static constexpr int HD   = 64;
static constexpr int MTOK = BB * LL;      // 4096 tokens
static constexpr int NQGKV = 4 * DOUT;    // 4096: [q | g | k | v]

#if defined(__has_builtin)
#if __has_builtin(__builtin_amdgcn_tensor_load_to_lds) && \
    __has_builtin(__builtin_amdgcn_s_wait_tensorcnt)
#define USE_TDM 1
#endif
#endif
#ifndef USE_TDM
#define USE_TDM 0
#endif

__device__ inline f8 zero8() {
  f8 z;
  for (int i = 0; i < 8; ++i) z[i] = 0.0f;
  return z;
}

// A/B fragment loader for V_WMMA_F32_16X16X32_F16 from a K-major matrix
// (global or LDS). Lane lo = lane&15 owns row `row` (M for A, N for B);
// lane hi = lane>>4 selects K sub-chunks hi*8+{0..7} and 16+hi*8+{0..7}.
__device__ inline h16 load_frag(const _Float16* __restrict__ base, int ld,
                                int row, int k0, int hi) {
  const _Float16* p = base + (size_t)row * (size_t)ld + k0 + hi * 8;
  h8 a = *(const h8*)p;
  h8 b = *(const h8*)(p + 16);
  return __builtin_shufflevector(a, b, 0, 1, 2, 3, 4, 5, 6, 7,
                                 8, 9, 10, 11, 12, 13, 14, 15);
}

__device__ inline f8 wmma16(h16 a, h16 b, f8 c) {
  // D = A(16x32 f16) x B(32x16 f16) + C(16x16 f32)
  return __builtin_amdgcn_wmma_f32_16x16x32_f16(
      false, a, false, b, (short)0, c, false, false);
}

#if USE_TDM
// Issue one TDM 2-D tile load: 128 rows x 64 halves (row-packed into LDS),
// source row stride = stride_halves. D# built per CDNA5 ISA 08 §8.3/8.4.
__device__ inline void tdm_load_2d(unsigned lds_off,
                                   const _Float16* __restrict__ gsrc,
                                   int stride_halves) {
  unsigned long long ga = (unsigned long long)gsrc;
  u32x4 g0;
  g0[0] = 1u;                                          // count=1, user D#
  g0[1] = lds_off;                                     // lds_addr (bytes)
  g0[2] = (unsigned)ga;                                // global_addr[31:0]
  g0[3] = (unsigned)((ga >> 32) & 0x01ffffffu)         // global_addr[56:32]
          | (2u << 30);                                // type=2 ("image")
  i32x8 g1;
  g1[0] = 1 << 16;                                     // mask=0, data_size=2B
  g1[1] = (stride_halves & 0xffff) << 16;              // tensor_dim0[15:0]
  g1[2] = ((stride_halves >> 16) & 0xffff)             // tensor_dim0[31:16]
          | (128 << 16);                               // tensor_dim1[15:0]=128
  g1[3] = (64 << 16);                                  // tile_dim0=64
  g1[4] = 128;                                         // tile_dim1=128, dim2=0
  g1[5] = stride_halves;                               // dim0_stride[31:0]
  g1[6] = 0;                                           // stride hi, dim1_stride
  g1[7] = 0;
  i32x4 z4 = {0, 0, 0, 0};
#if defined(__clang_major__) && __clang_major__ >= 23
  i32x8 z8 = {0, 0, 0, 0, 0, 0, 0, 0};
  __builtin_amdgcn_tensor_load_to_lds(g0, g1, z4, z4, z8, 0);
#else
  __builtin_amdgcn_tensor_load_to_lds(g0, g1, z4, z4, 0);
#endif
}
#else
// Fallback staging: cooperative vector copy of a 128x64-half tile.
__device__ inline void stage_tile(_Float16* __restrict__ dst,
                                  const _Float16* __restrict__ src,
                                  int stride_halves, int tid) {
  for (int c = tid; c < 1024; c += 256) {  // 1024 chunks of 8 halves
    int row = c >> 3, col = (c & 7) * 8;
    *(h8*)(dst + row * 64 + col) =
        *(const h8*)(src + (size_t)row * stride_halves + col);
  }
}
#endif

// ---------------------------------------------------------------------------
// f32 -> f16 convert (4 elems / thread, sizes are multiples of 1024)
// ---------------------------------------------------------------------------
__global__ void cvt_f32_f16(const float* __restrict__ src,
                            _Float16* __restrict__ dst, int n) {
  int i = (blockIdx.x * blockDim.x + threadIdx.x) * 4;
  if (i < n) {
    float4 v = *(const float4*)(src + i);
    h4 o = {(_Float16)v.x, (_Float16)v.y, (_Float16)v.z, (_Float16)v.w};
    *(h4*)(dst + i) = o;
  }
}

// ---------------------------------------------------------------------------
// WMMA GEMM:  Out[M][N] = A[M][K] * Bm[N][K]^T   (i.e. x @ W^T)
// Block: 256 threads = 8 waves as 2(M) x 4(N); block tile 128x128;
// wave tile 64(M) x 32(N). A and B K-slabs (128x64 halves, 16 KB each)
// are DMA'd into LDS by the Tensor Data Mover (wave 0), all waves consume
// fragments via ds_load_b128. FINAL=1: f32 out + bias; FINAL=0: f16 out.
// ---------------------------------------------------------------------------
template <int FINAL>
__global__ __launch_bounds__(256) void gemm_wmma(
    const _Float16* __restrict__ A, const _Float16* __restrict__ Bm,
    void* __restrict__ Outv, const float* __restrict__ bias,
    int M, int N, int K) {
  __shared__ __align__(16) _Float16 ash[128][64];
  __shared__ __align__(16) _Float16 bsh[128][64];

  int w = threadIdx.x >> 5, lane = threadIdx.x & 31;
  int lo = lane & 15, hi = lane >> 4;
  int wm = w >> 2, wn = w & 3;
  int mblk = blockIdx.y * 128, nblk = blockIdx.x * 128;
  int mw = wm * 64, nw = wn * 32;

  f8 acc[4][2];
  for (int i = 0; i < 4; ++i)
    for (int j = 0; j < 2; ++j) acc[i][j] = zero8();

  for (int k0 = 0; k0 < K; k0 += 64) {
#if USE_TDM
    if (threadIdx.x < 32) {  // wave 0 drives the TDM, waits TENSORcnt==0
      tdm_load_2d((unsigned)(size_t)&ash[0][0], A + (size_t)mblk * K + k0, K);
      tdm_load_2d((unsigned)(size_t)&bsh[0][0], Bm + (size_t)nblk * K + k0, K);
      __builtin_amdgcn_s_wait_tensorcnt(0);
    }
#else
    stage_tile(&ash[0][0], A + (size_t)mblk * K + k0, K, threadIdx.x);
    stage_tile(&bsh[0][0], Bm + (size_t)nblk * K + k0, K, threadIdx.x);
#endif
    __syncthreads();

#pragma unroll
    for (int kk = 0; kk < 64; kk += 32) {
      h16 af[4], bf[2];
#pragma unroll
      for (int i = 0; i < 4; ++i)
        af[i] = load_frag(&ash[0][0], 64, mw + i * 16 + lo, kk, hi);
#pragma unroll
      for (int j = 0; j < 2; ++j)
        bf[j] = load_frag(&bsh[0][0], 64, nw + j * 16 + lo, kk, hi);
#pragma unroll
      for (int i = 0; i < 4; ++i)
#pragma unroll
        for (int j = 0; j < 2; ++j) acc[i][j] = wmma16(af[i], bf[j], acc[i][j]);
    }
    __syncthreads();
  }

  // C layout: VGPR v -> row mblk+mw+i*16 + v + hi*8, col nblk+nw+j*16 + lo
#pragma unroll
  for (int i = 0; i < 4; ++i)
#pragma unroll
    for (int j = 0; j < 2; ++j)
#pragma unroll
      for (int v = 0; v < 8; ++v) {
        int row = mblk + mw + i * 16 + v + hi * 8;
        int col = nblk + nw + j * 16 + lo;
        float val = acc[i][j][v];
        if constexpr (FINAL) {
          ((float*)Outv)[(size_t)row * N + col] = val + bias[col];
        } else {
          ((_Float16*)Outv)[(size_t)row * N + col] = (_Float16)val;
        }
      }
}

// ---------------------------------------------------------------------------
// Build V^T per head: vt[(b*H+h)*64 + d][t] = qgkv[(b*L+t)][3072 + h*64 + d]
// ---------------------------------------------------------------------------
__global__ void build_vt(const _Float16* __restrict__ qgkv,
                         _Float16* __restrict__ vt) {
  int idx = blockIdx.x * blockDim.x + threadIdx.x;  // [B*H*HD][L], t fastest
  int t = idx & (LL - 1);
  int r = idx >> 11;  // (b*16+h)*64 + d
  int d = r & 63;
  int bh = r >> 6;
  int h = bh & 15, b = bh >> 4;
  vt[(size_t)idx] =
      qgkv[((size_t)(b * LL + t)) * NQGKV + 3 * DOUT + h * HD + d];
}

// ---------------------------------------------------------------------------
// Causal flash attention + sigmoid(g) gate.
// Grid: (B*H, L/64); block 128 = 4 waves, each wave owns 16 query rows.
// S = Q K^T (WMMA), online softmax (shfl_xor row reductions), P via LDS
// C-layout -> A-layout round trip, ctx += P V (WMMA, V^T rows contiguous).
// ---------------------------------------------------------------------------
__global__ __launch_bounds__(128) void attn_flash(
    const _Float16* __restrict__ qgkv, const _Float16* __restrict__ vt,
    _Float16* __restrict__ ctxout) {
  __shared__ __align__(16) _Float16 plds[4][16][64];  // per-wave P tile

  int bh = blockIdx.x;
  int b = bh >> 4, h = bh & 15;
  int qb = blockIdx.y;
  int w = threadIdx.x >> 5, lane = threadIdx.x & 31;
  int lo = lane & 15, hi = lane >> 4;

  const _Float16* Qb = qgkv + (size_t)b * LL * NQGKV + h * HD;  // q section
  const _Float16* Gb = Qb + DOUT;                               // g section
  const _Float16* Kb = Qb + 2 * DOUT;                           // k section
  const _Float16* Vb = vt + (size_t)bh * HD * LL;               // V^T rows

  int q0 = qb * 64 + w * 16;  // this wave's 16 query rows

  h16 qf[2];
  qf[0] = load_frag(Qb, NQGKV, q0 + lo, 0, hi);
  qf[1] = load_frag(Qb, NQGKV, q0 + lo, 32, hi);

  f8 cacc[4];
  for (int j = 0; j < 4; ++j) cacc[j] = zero8();
  float rm[8], rl[8];
  for (int v = 0; v < 8; ++v) { rm[v] = -3.0e38f; rl[v] = 0.0f; }

  for (int kb = 0; kb <= qb; ++kb) {
    int k0 = kb * 64;
    f8 s[4];
#pragma unroll
    for (int j = 0; j < 4; ++j) {
      s[j] = zero8();
      h16 kf0 = load_frag(Kb, NQGKV, k0 + j * 16 + lo, 0, hi);
      h16 kf1 = load_frag(Kb, NQGKV, k0 + j * 16 + lo, 32, hi);
      s[j] = wmma16(qf[0], kf0, s[j]);
      s[j] = wmma16(qf[1], kf1, s[j]);
    }

    bool diag = (kb == qb);
#pragma unroll
    for (int v = 0; v < 8; ++v) {
      int qrow = q0 + v + hi * 8;
      float mx = -3.0e38f;
#pragma unroll
      for (int j = 0; j < 4; ++j) {
        float val = s[j][v] * 0.125f;  // 1/sqrt(HD)
        if (diag && (k0 + j * 16 + lo) > qrow) val = -3.0e38f;  // causal mask
        s[j][v] = val;
        mx = fmaxf(mx, val);
      }
      for (int m = 1; m < 16; m <<= 1) mx = fmaxf(mx, __shfl_xor(mx, m, 32));
      float nm = fmaxf(rm[v], mx);
      float alpha = __expf(rm[v] - nm);
      float rs = 0.0f;
#pragma unroll
      for (int j = 0; j < 4; ++j) {
        float p = __expf(s[j][v] - nm);
        s[j][v] = p;
        rs += p;
      }
      for (int m = 1; m < 16; m <<= 1) rs += __shfl_xor(rs, m, 32);
      rl[v] = rl[v] * alpha + rs;
      rm[v] = nm;
#pragma unroll
      for (int j = 0; j < 4; ++j) {
        cacc[j][v] *= alpha;
        plds[w][v + hi * 8][j * 16 + lo] = (_Float16)s[j][v];  // C-layout scatter
      }
    }

    // Re-read P as A-fragments (per-wave LDS slab; DS ops in-order in wave).
    const _Float16* prow = &plds[w][lo][0];
    h8 pa0 = *(const h8*)(prow + hi * 8);
    h8 pb0 = *(const h8*)(prow + 16 + hi * 8);
    h8 pa1 = *(const h8*)(prow + 32 + hi * 8);
    h8 pb1 = *(const h8*)(prow + 48 + hi * 8);
    h16 pf0 = __builtin_shufflevector(pa0, pb0, 0, 1, 2, 3, 4, 5, 6, 7,
                                      8, 9, 10, 11, 12, 13, 14, 15);
    h16 pf1 = __builtin_shufflevector(pa1, pb1, 0, 1, 2, 3, 4, 5, 6, 7,
                                      8, 9, 10, 11, 12, 13, 14, 15);

#pragma unroll
    for (int j = 0; j < 4; ++j) {
      h16 vf0 = load_frag(Vb, LL, j * 16 + lo, k0, hi);
      h16 vf1 = load_frag(Vb, LL, j * 16 + lo, k0 + 32, hi);
      cacc[j] = wmma16(pf0, vf0, cacc[j]);
      cacc[j] = wmma16(pf1, vf1, cacc[j]);
    }
  }

  // normalize, gate with sigmoid(g), store f16 ctx [token][h*64+d]
#pragma unroll
  for (int j = 0; j < 4; ++j)
#pragma unroll
    for (int v = 0; v < 8; ++v) {
      int qrow = q0 + v + hi * 8;
      int d = j * 16 + lo;
      float g = (float)Gb[(size_t)qrow * NQGKV + d];
      float gate = 1.0f / (1.0f + __expf(-g));
      float val = (cacc[j][v] / rl[v]) * gate;
      ctxout[((size_t)(b * LL + qrow)) * DOUT + h * HD + d] = (_Float16)val;
    }
}

// ---------------------------------------------------------------------------
extern "C" void kernel_launch(void* const* d_in, const int* in_sizes, int n_in,
                              void* d_out, int out_size, void* d_ws,
                              size_t ws_size, hipStream_t stream) {
  (void)in_sizes; (void)n_in; (void)out_size; (void)ws_size;
  const float* x  = (const float*)d_in[0];
  const float* Wq = (const float*)d_in[1];
  const float* Wg = (const float*)d_in[2];
  const float* Wk = (const float*)d_in[3];
  const float* Wv = (const float*)d_in[4];
  const float* Wo = (const float*)d_in[5];
  const float* bo = (const float*)d_in[6];
  float* out = (float*)d_out;

  _Float16* ws   = (_Float16*)d_ws;
  _Float16* xh   = ws;                               // 4096*1024
  _Float16* Wh   = xh + (size_t)MTOK * DIN;          // 4096*1024 fused W
  _Float16* Woh  = Wh + (size_t)NQGKV * DIN;         // 1024*1024
  _Float16* qgkv = Woh + (size_t)DOUT * DOUT;        // 4096*4096
  _Float16* vtb  = qgkv + (size_t)MTOK * NQGKV;      // 2048*2048
  _Float16* ctx  = vtb + (size_t)BB * HH * HD * LL;  // 4096*1024

  const int WN = DIN * DOUT;  // 1M elements per weight
  cvt_f32_f16<<<(MTOK * DIN) / 1024, 256, 0, stream>>>(x, xh, MTOK * DIN);
  cvt_f32_f16<<<WN / 1024, 256, 0, stream>>>(Wq, Wh + 0 * (size_t)WN, WN);
  cvt_f32_f16<<<WN / 1024, 256, 0, stream>>>(Wg, Wh + 1 * (size_t)WN, WN);
  cvt_f32_f16<<<WN / 1024, 256, 0, stream>>>(Wk, Wh + 2 * (size_t)WN, WN);
  cvt_f32_f16<<<WN / 1024, 256, 0, stream>>>(Wv, Wh + 3 * (size_t)WN, WN);
  cvt_f32_f16<<<WN / 1024, 256, 0, stream>>>(Wo, Woh, WN);

  // QGKV = xh @ Wh^T : M=4096, N=4096, K=1024
  gemm_wmma<0><<<dim3(NQGKV / 128, MTOK / 128), 256, 0, stream>>>(
      xh, Wh, qgkv, nullptr, MTOK, NQGKV, DIN);

  build_vt<<<(BB * HH * HD * LL) / 256, 256, 0, stream>>>(qgkv, vtb);

  attn_flash<<<dim3(BB * HH, LL / 64), 128, 0, stream>>>(qgkv, vtb, ctx);

  // out = ctx @ Wo^T + bo : M=4096, N=1024, K=1024, f32
  gemm_wmma<1><<<dim3(DOUT / 128, MTOK / 128), 256, 0, stream>>>(
      ctx, Woh, out, bo, MTOK, DOUT, DOUT);
}
